// CapsuleMortalityHead_61014305407747
// MI455X (gfx1250) — matrix operand order repaired and checked by reading, not compiled, not run on hardware
//
#include <hip/hip_runtime.h>
#include <hip/hip_bf16.h>
#include <stdint.h>

typedef __attribute__((ext_vector_type(16))) __bf16 v16bf;
typedef __attribute__((ext_vector_type(8)))  float  v8f;
typedef __attribute__((ext_vector_type(4)))  int    v4i;

#define B_TOT   4096
#define N_IN    7
#define PC_DIM  256
#define KCAP    25
#define MC_DIM  128
#define CVEC    (KCAP*MC_DIM)   /* 3200 */
#define NK      (N_IN*KCAP)     /* 175  */
#define LN_EPS  1e-5f

#define ROWT    (B_TOT/16)      /* 256 row tiles  */
#define COLT    (CVEC/16)       /* 200 col tiles  */
#define DCH     (PC_DIM/32)     /* 8 d-chunks     */

#if __has_builtin(__builtin_amdgcn_global_load_async_to_lds_b128)
#define HAVE_ASYNC_LDS 1
#else
#define HAVE_ASYNC_LDS 0
#endif

typedef __attribute__((address_space(1))) v4i* gv4i_p;   // global
typedef __attribute__((address_space(3))) v4i* lv4i_p;   // LDS

// A-fragment K index for 16-bit WMMA A matrix (ISA 7.12.2)
__device__ __forceinline__ int a_kidx(int i, int half) {
    return half * 8 + (i < 8 ? i : i + 8);
}

// ==================================================================
// Pack kernels: f32 -> bf16, rearranged into per-wave fragment order
// A'[bt][n][d0][lane][i] ; W'[n][kct][d0][lane][i]  (16 bf16 = 32 B/lane)
// ==================================================================
__global__ __launch_bounds__(256)
void pack_poses(const float* __restrict__ poses, __bf16* __restrict__ Ap)
{
    int tid  = blockIdx.x * 256 + threadIdx.x;
    int i    = tid & 15;
    int lane = (tid >> 4) & 31;
    int rest = tid >> 9;
    int d0   = rest & 7;  rest >>= 3;
    int n    = rest % N_IN;
    int bt   = rest / N_IN;
    int b    = bt * 16 + (lane & 15);
    int d    = d0 * 32 + a_kidx(i, lane >> 4);
    Ap[tid] = (__bf16)poses[((size_t)b * N_IN + n) * PC_DIM + d];
}

__global__ __launch_bounds__(256)
void pack_w(const float* __restrict__ w, __bf16* __restrict__ Wp)
{
    int tid  = blockIdx.x * 256 + threadIdx.x;
    int i    = tid & 15;
    int lane = (tid >> 4) & 31;
    int rest = tid >> 9;
    int d0   = rest & 7;  rest >>= 3;
    int kct  = rest % COLT;
    int n    = rest / COLT;
    int kc   = kct * 16 + (lane & 15);
    int d    = d0 * 32 + a_kidx(i, lane >> 4);          // B: K=d, N=kc
    Wp[tid] = (__bf16)w[((size_t)n * PC_DIM + d) * CVEC + kc];
}

// ==================================================================
// Kernel 1: votes GEMM. Each wave: 32x32 macro-tile (2x2 WMMA tiles),
// fragments loaded as contiguous v16bf (b128 pairs) from packed arrays.
// total waves = 7 * 128 rowpairs * 100 colpairs = 89600 ; 8 waves/block.
// ==================================================================
__global__ __launch_bounds__(256)
void votes_gemm_bf16(const __bf16* __restrict__ Ap,
                     const __bf16* __restrict__ Wp,
                     __bf16* __restrict__ votes)
{
    const int lane = threadIdx.x & 31;
    const int wid  = threadIdx.x >> 5;
    const int Wg   = blockIdx.x * 8 + wid;

    const int n   = Wg / ((ROWT/2) * (COLT/2));
    const int rem = Wg % ((ROWT/2) * (COLT/2));
    const int rp  = rem / (COLT/2);      // row-tile pair
    const int cp  = rem % (COLT/2);      // col-tile pair

    const int half = lane >> 4;
    const int l16  = lane & 15;

    const v16bf* A16 = (const v16bf*)Ap;   // 512 bf16 per (bt,n,d0) = 32 v16bf
    const v16bf* B16 = (const v16bf*)Wp;

    v8f acc00 = {}, acc01 = {}, acc10 = {}, acc11 = {};

    #pragma unroll
    for (int d0 = 0; d0 < DCH; ++d0) {
        v16bf a0 = A16[(size_t)(((rp*2 + 0) * N_IN + n) * DCH + d0) * 32 + lane];
        v16bf a1 = A16[(size_t)(((rp*2 + 1) * N_IN + n) * DCH + d0) * 32 + lane];
        v16bf b0 = B16[(size_t)((n * COLT + cp*2 + 0) * DCH + d0) * 32 + lane];
        v16bf b1 = B16[(size_t)((n * COLT + cp*2 + 1) * DCH + d0) * 32 + lane];
        acc00 = __builtin_amdgcn_wmma_f32_16x16x32_bf16(false, a0, false, b0, (short)0, acc00, false, false);
        acc01 = __builtin_amdgcn_wmma_f32_16x16x32_bf16(false, a0, false, b1, (short)0, acc01, false, false);
        acc10 = __builtin_amdgcn_wmma_f32_16x16x32_bf16(false, a1, false, b0, (short)0, acc10, false, false);
        acc11 = __builtin_amdgcn_wmma_f32_16x16x32_bf16(false, a1, false, b1, (short)0, acc11, false, false);
    }

    // C layout: VGPR i -> M = i + 8*half, N = l16
    #pragma unroll
    for (int r = 0; r < 2; ++r) {
        #pragma unroll
        for (int c = 0; c < 2; ++c) {
            const v8f acc = r ? (c ? acc11 : acc10) : (c ? acc01 : acc00);
            #pragma unroll
            for (int i = 0; i < 8; ++i) {
                int row = (rp*2 + r) * 16 + i + 8*half;
                size_t idx = ((size_t)row * N_IN + n) * CVEC + (cp*2 + c) * 16 + l16;
                votes[idx] = (__bf16)acc[i];
            }
        }
    }
}

// ==================================================================
// Kernel 2: full routing for one batch row per 128-thread block.
// votes row kept bf16 in LDS (44.8 KB), filled via async-to-LDS.
// ==================================================================
__device__ __forceinline__ float wred32(float v) {
    #pragma unroll
    for (int off = 16; off > 0; off >>= 1)
        v += __shfl_xor(v, off, 32);
    return v;
}

__device__ __forceinline__ void layernorm128(float* pose, float gm, float bm,
                                             int lane, int wv, int t,
                                             float* red, float* mu_s, float* rs_s)
{
    #pragma unroll
    for (int k = 0; k < KCAP; ++k) {
        float s1 = wred32(pose[k]);
        float s2 = wred32(pose[k] * pose[k]);
        if (lane == 0) { red[wv*KCAP + k] = s1; red[(4 + wv)*KCAP + k] = s2; }
    }
    __syncthreads();
    if (t < KCAP) {
        float s1 = red[t] + red[KCAP+t] + red[2*KCAP+t] + red[3*KCAP+t];
        float s2 = red[4*KCAP+t] + red[5*KCAP+t] + red[6*KCAP+t] + red[7*KCAP+t];
        float muv = s1 * (1.0f / 128.0f);
        float var = s2 * (1.0f / 128.0f) - muv * muv;
        mu_s[t] = muv;
        rs_s[t] = rsqrtf(var + LN_EPS);
    }
    __syncthreads();
    #pragma unroll
    for (int k = 0; k < KCAP; ++k)
        pose[k] = (pose[k] - mu_s[k]) * rs_s[k] * gm + bm;
    __syncthreads();
}

__global__ __launch_bounds__(128)
void routing_kernel(const __bf16* __restrict__ votes,
                    const float* __restrict__ acts,
                    const float* __restrict__ gamma,
                    const float* __restrict__ beta,
                    const float* __restrict__ clsw,
                    const float* __restrict__ clsb,
                    float* __restrict__ out_logits,
                    float* __restrict__ out_r)
{
    __shared__ __attribute__((aligned(16))) __bf16 vlds[N_IN * CVEC]; // 44.8 KB
    __shared__ float red[4 * NK];
    __shared__ float rlds[NK];
    __shared__ float ralds[NK];
    __shared__ float mu_s[KCAP], rs_s[KCAP];
    __shared__ float a_s[N_IN];

    const int t    = threadIdx.x;
    const int m    = t;
    const int lane = t & 31;
    const int wv   = t >> 5;
    const int b    = blockIdx.x;

    // ---- fill votes row into LDS (44800 B = 2800 x b128) ----
    {
        const uint4* gsrc = (const uint4*)(votes + (size_t)b * (N_IN * CVEC));
        uint4* ldst = (uint4*)vlds;
#if HAVE_ASYNC_LDS
        for (int c = t; c < 2800; c += 128) {
            __builtin_amdgcn_global_load_async_to_lds_b128(
                (gv4i_p)(gsrc + c), (lv4i_p)(ldst + c), 0, 0);
        }
#if __has_builtin(__builtin_amdgcn_s_wait_asynccnt)
        __builtin_amdgcn_s_wait_asynccnt(0);
#else
        asm volatile("s_wait_asynccnt 0" ::: "memory");
#endif
#else
        for (int c = t; c < 2800; c += 128)
            ldst[c] = gsrc[c];
#endif
    }
    if (t < N_IN) a_s[t] = acts[b * N_IN + t];
    __syncthreads();

    const float gm = gamma[m];
    const float bm = beta[m];
    const float scale = 0.08838834764831845f;   // 1/sqrt(128)

    float pose[KCAP];

    // ---- iteration 0: uniform routing ----
    #pragma unroll
    for (int k = 0; k < KCAP; ++k) {
        float s = 0.f;
        #pragma unroll
        for (int n = 0; n < N_IN; ++n)
            s += (a_s[n] * (1.0f / (float)KCAP)) * (float)vlds[(n*KCAP + k)*MC_DIM + m];
        pose[k] = s;
    }
    layernorm128(pose, gm, bm, lane, wv, t, red, mu_s, rs_s);

    // ---- iterations 1..2: agreement routing ----
    for (int it = 1; it < 3; ++it) {
        for (int n = 0; n < N_IN; ++n) {
            #pragma unroll
            for (int k = 0; k < KCAP; ++k) {
                float s = wred32((float)vlds[(n*KCAP + k)*MC_DIM + m] * pose[k]);
                if (lane == 0) red[wv*NK + n*KCAP + k] = s;
            }
        }
        __syncthreads();
        if (t < NK)
            rlds[t] = (red[t] + red[NK + t] + red[2*NK + t] + red[3*NK + t]) * scale;
        __syncthreads();
        if (t < N_IN) {
            float mx = -1e30f;
            for (int k = 0; k < KCAP; ++k) mx = fmaxf(mx, rlds[t*KCAP + k]);
            float sum = 0.f;
            for (int k = 0; k < KCAP; ++k) sum += __expf(rlds[t*KCAP + k] - mx);
            float inv = 1.0f / sum;
            float an  = a_s[t];
            for (int k = 0; k < KCAP; ++k) {
                float rv = __expf(rlds[t*KCAP + k] - mx) * inv;
                rlds[t*KCAP + k]  = rv;
                ralds[t*KCAP + k] = rv * an;
            }
        }
        __syncthreads();
        #pragma unroll
        for (int k = 0; k < KCAP; ++k) {
            float s = 0.f;
            #pragma unroll
            for (int n = 0; n < N_IN; ++n)
                s += ralds[n*KCAP + k] * (float)vlds[(n*KCAP + k)*MC_DIM + m];
            pose[k] = s;
        }
        layernorm128(pose, gm, bm, lane, wv, t, red, mu_s, rs_s);
    }

    // ---- classifier ----
    float cw[KCAP];
    #pragma unroll
    for (int k = 0; k < KCAP; ++k) cw[k] = clsw[k * MC_DIM + m];
    #pragma unroll
    for (int k = 0; k < KCAP; ++k) {
        float s = wred32(pose[k] * cw[k]);
        if (lane == 0) red[wv*KCAP + k] = s;
    }
    __syncthreads();
    if (t < KCAP)
        out_logits[(size_t)b * KCAP + t] =
            red[t] + red[KCAP + t] + red[2*KCAP + t] + red[3*KCAP + t] + clsb[t];
    if (t < NK)
        out_r[(size_t)b * NK + t] = rlds[t];
}

// ==================================================================
extern "C" void kernel_launch(void* const* d_in, const int* in_sizes, int n_in,
                              void* d_out, int out_size, void* d_ws, size_t ws_size,
                              hipStream_t stream)
{
    const float* poses = (const float*)d_in[0];
    const float* acts  = (const float*)d_in[1];
    const float* w     = (const float*)d_in[2];
    const float* g     = (const float*)d_in[3];
    const float* bta   = (const float*)d_in[4];
    const float* cw    = (const float*)d_in[5];
    const float* cb    = (const float*)d_in[6];
    float* out = (float*)d_out;

    // ws layout (bf16): votes | packed A | packed W
    char* ws = (char*)d_ws;
    __bf16* votes = (__bf16*)ws;                               // 4096*22400 = 183.5 MB
    __bf16* Ap    = (__bf16*)(ws + (size_t)B_TOT * N_IN * CVEC * 2);        // 14.68 MB
    __bf16* Wp    = (__bf16*)(ws + (size_t)B_TOT * N_IN * CVEC * 2
                                 + (size_t)B_TOT * N_IN * PC_DIM * 2);      // 11.47 MB

    pack_poses<<<dim3((B_TOT*N_IN*PC_DIM)/256), 256, 0, stream>>>(poses, Ap);
    pack_w<<<dim3((N_IN*PC_DIM*CVEC)/256), 256, 0, stream>>>(w, Wp);

    // 7 * (256/2) * (200/2) = 89600 waves, 8 waves per block
    votes_gemm_bf16<<<dim3(89600/8), 256, 0, stream>>>(Ap, Wp, votes);

    routing_kernel<<<dim3(B_TOT), 128, 0, stream>>>(
        votes, acts, g, bta, cw, cb,
        out, out + (size_t)B_TOT * KCAP);
}
